// ContrastiveGNN_44100724196062
// MI455X (gfx1250) — compile-verified, matrix operand order checked by reading, not run on hardware
//
#include <hip/hip_runtime.h>

// ---------------------------------------------------------------------------
// Problem constants (match reference)
// ---------------------------------------------------------------------------
#define NP_   50000
#define NS_   20000
#define NT_   5000
#define NTOT  75000
#define E_PS  400000
#define E_ST  100000
#define ETOT  500000
#define D_IN  256
#define D_H   128
#define HEADS 4
#define CPH   32           // channels per head
#define LN_EPS   1e-5f
#define SM_EPS   1e-16f
#define INV_SQRT_C 0.17677669529663687f   // 1/sqrt(32)

#define MTILE 64           // rows per block in the WMMA GEMM

// ---------------------------------------------------------------------------
// WMMA types (CDNA5 wave32, V_WMMA_F32_16X16X32_BF16)
// ---------------------------------------------------------------------------
typedef __attribute__((ext_vector_type(16))) __bf16 v16bf;
typedef __attribute__((ext_vector_type(8)))  float  v8f;
typedef __attribute__((ext_vector_type(8)))  unsigned short us8;

union Frag {
    v16bf v;
    us8   h[2];
    unsigned short u[16];
};

__device__ __forceinline__ unsigned short f2bf(float f) {
    unsigned u = __float_as_uint(f);
    unsigned r = u + 0x7fffu + ((u >> 16) & 1u);   // round to nearest even
    return (unsigned short)(r >> 16);
}

// monotonic float<->uint encoding for atomicMax-based segment max
__device__ __forceinline__ unsigned fenc(float f) {
    unsigned u = __float_as_uint(f);
    return (u & 0x80000000u) ? ~u : (u | 0x80000000u);
}
__device__ __forceinline__ float fdec(unsigned e) {
    unsigned u = (e & 0x80000000u) ? (e & 0x7fffffffu) : ~e;
    return __uint_as_float(u);
}
#define ENC_NEG_INF 0x007FFFFFu   // fenc(-inf)

// ---------------------------------------------------------------------------
// Weight pack: f32 [K,128] row-major  ->  bf16 in WMMA B-fragment order.
// Packed index = (((t*8 + wv)*32 + lane) * 16 + e), where
//   t = k-tile (k0/32), wv = 16-col tile, lane = wave lane, e = frag element.
// Lane's 16 values are contiguous (32B) -> B frag = two global_load_b128.
// ---------------------------------------------------------------------------
__global__ void pack_weight_bf16(const float* __restrict__ W,
                                 unsigned short* __restrict__ dst, int K) {
    int idx = blockIdx.x * 256 + threadIdx.x;
    if (idx >= K * 128) return;
    int e    = idx & 15;
    int lane = (idx >> 4) & 31;
    int wvt  = idx >> 9;          // t*8 + wv
    int wv   = wvt & 7;
    int t    = wvt >> 3;
    int hf   = lane >> 4;
    int n    = (wv << 4) | (lane & 15);
    int k    = (t << 5) + ((e < 8) ? (e + 8 * hf) : (e + 8 + 8 * hf));
    dst[idx] = f2bf(W[(size_t)k * 128 + n]);
}

// ---------------------------------------------------------------------------
// Multi-matrix GEMM: C[m][M,128] = A[M,K](f32) * W[m][K,128](bf16, packed) + b
// block: 256 threads = 8 waves; block covers 64 rows x 128 cols.
// A staged once in LDS (bf16), reused by all nmat weight matrices (q/k/v/skip).
// Per k-step each wave: 2 global_load_b128 (B) + 4 x (2 ds_load_b128 + wmma).
// ---------------------------------------------------------------------------
struct GemmArgs {
    const float*          A;
    const unsigned short* W[4];
    const float*          bias[4];
    float*                C[4];
    int M, K, nmat;
};

__global__ __launch_bounds__(256)
void gemm_multi(GemmArgs args) {
    __shared__ __align__(16) unsigned short As[MTILE * 256];   // up to K=256 (32KB)

    const int tid  = threadIdx.x;
    const int row0 = blockIdx.x * MTILE;
    const int M = args.M, K = args.K;

    // stage A tile (f32 -> bf16) into LDS, coalesced along K (K is 128 or 256)
    {
        int kk    = tid & (K - 1);
        int rstep = 256 / K;          // 1 or 2
        int rbase = tid / K;
        for (int r = rbase; r < MTILE; r += rstep) {
            int gr  = row0 + r;
            float v = (gr < M) ? args.A[(size_t)gr * K + kk] : 0.0f;
            As[r * K + kk] = f2bf(v);
        }
    }
    __syncthreads();

    const int lane = tid & 31;
    const int wv   = tid >> 5;        // 0..7 -> 16-col N tile
    const int r    = lane & 15;
    const int hf   = lane >> 4;
    const int n    = (wv << 4) | r;   // global column 0..127

    for (int mat = 0; mat < args.nmat; ++mat) {
        const unsigned short* Wp = args.W[mat];
        v8f acc[4] = {};

        for (int k0 = 0; k0 < K; k0 += 32) {
            const int t = k0 >> 5;
            // B fragment: packed layout -> 32 contiguous bytes per lane
            const unsigned short* bp = Wp + ((((size_t)t * 8 + wv) * 32 + lane) << 4);
            Frag b;
            b.h[0] = *(const us8*)bp;
            b.h[1] = *(const us8*)(bp + 8);
#pragma unroll
            for (int mt = 0; mt < 4; ++mt) {
                Frag a;
                const int ar = (mt << 4) | r;
                a.h[0] = *(const us8*)&As[ar * K + k0 + 8 * hf];
                a.h[1] = *(const us8*)&As[ar * K + k0 + 16 + 8 * hf];
                acc[mt] = __builtin_amdgcn_wmma_f32_16x16x32_bf16(
                              false, a.v, false, b.v, (short)0, acc[mt], false, false);
            }
        }

        const float bv = args.bias[mat][n];
        float* Cm = args.C[mat];
#pragma unroll
        for (int mt = 0; mt < 4; ++mt) {
#pragma unroll
            for (int j = 0; j < 8; ++j) {
                int m = row0 + (mt << 4) + j + 8 * hf;   // C layout: VGPR j -> row j+8*hf
                if (m < M) Cm[(size_t)m * 128 + n] = acc[mt][j] + bv;
            }
        }
    }
}

// ---------------------------------------------------------------------------
// per-layer scratch init
// ---------------------------------------------------------------------------
__global__ void init_layer(float* __restrict__ accum, float* __restrict__ denom,
                           unsigned* __restrict__ amax, int nAccum, int nND) {
    int i = blockIdx.x * 256 + threadIdx.x;
    if (i < nAccum) accum[i] = 0.0f;
    if (i < nND) { denom[i] = 0.0f; amax[i] = ENC_NEG_INF; }
}

// ---------------------------------------------------------------------------
// edge helpers: resolve heterogeneous edge -> global (src, dst, attr)
// ---------------------------------------------------------------------------
__device__ __forceinline__ void edge_resolve(int e,
        const long long* __restrict__ ps_ei, const long long* __restrict__ st_ei,
        const float* __restrict__ ps_attr, int& s, int& d, float& attr) {
    if (e < E_PS) {
        s = (int)ps_ei[e];
        d = (int)ps_ei[E_PS + e] + NP_;
        attr = ps_attr[e];
    } else {
        int e2 = e - E_PS;
        s = (int)st_ei[e2] + NP_;
        d = (int)st_ei[E_ST + e2] + NP_ + NS_;
        attr = 1.0f;
    }
}

// pass 1: alpha[e,h] = <q[dst], k[src]+edge> / sqrt(C);  segment max via atomicMax
__global__ __launch_bounds__(256)
void edge_pass1(const float* __restrict__ qb, const float* __restrict__ kb,
                const long long* __restrict__ ps_ei, const long long* __restrict__ st_ei,
                const float* __restrict__ ps_attr,
                const float* __restrict__ we, const float* __restrict__ be,
                float* __restrict__ alphab, unsigned* __restrict__ amax) {
    int e = blockIdx.x * 8 + (threadIdx.x >> 5);
    if (e >= ETOT) return;
    int lane = threadIdx.x & 31;

    int s, d; float attr;
    edge_resolve(e, ps_ei, st_ei, ps_attr, s, d, attr);

    const float* qd = qb + (size_t)d * 128;
    const float* ks = kb + (size_t)s * 128;

#pragma unroll
    for (int h = 0; h < HEADS; ++h) {
        int c = h * CPH + lane;
        float kj = ks[c] + attr * we[c] + be[c];
        float p  = qd[c] * kj;
#pragma unroll
        for (int o = 16; o > 0; o >>= 1) p += __shfl_xor(p, o, 32);
        if (lane == 0) {
            float al = p * INV_SQRT_C;
            alphab[(size_t)e * HEADS + h] = al;
            atomicMax(&amax[(size_t)d * HEADS + h], fenc(al));
        }
    }
}

// pass 2: a = exp(alpha - amax[dst]); denom[dst] += a; accum[dst] += a * (v[src]+edge)
__global__ __launch_bounds__(256)
void edge_pass2(const float* __restrict__ vb,
                const long long* __restrict__ ps_ei, const long long* __restrict__ st_ei,
                const float* __restrict__ ps_attr,
                const float* __restrict__ we, const float* __restrict__ be,
                const float* __restrict__ alphab, const unsigned* __restrict__ amax,
                float* __restrict__ denom, float* __restrict__ accum) {
    int e = blockIdx.x * 8 + (threadIdx.x >> 5);
    if (e >= ETOT) return;
    int lane = threadIdx.x & 31;

    int s, d; float attr;
    edge_resolve(e, ps_ei, st_ei, ps_attr, s, d, attr);

    const float* vs = vb + (size_t)s * 128;

#pragma unroll
    for (int h = 0; h < HEADS; ++h) {
        float am = fdec(amax[(size_t)d * HEADS + h]);
        float a  = __expf(alphab[(size_t)e * HEADS + h] - am);
        if (lane == 0)
            __hip_atomic_fetch_add(&denom[(size_t)d * HEADS + h], a,
                                   __ATOMIC_RELAXED, __HIP_MEMORY_SCOPE_AGENT);
        int c = h * CPH + lane;
        float vj = vs[c] + attr * we[c] + be[c];
        __hip_atomic_fetch_add(&accum[(size_t)d * 128 + c], a * vj,
                               __ATOMIC_RELAXED, __HIP_MEMORY_SCOPE_AGENT);
    }
}

// ---------------------------------------------------------------------------
// node update: out = accum/denom + skip; h += relu(LN(out))
// block = 128 threads (one node), 4 waves, cross-wave reduce via LDS
// ---------------------------------------------------------------------------
__global__ __launch_bounds__(128)
void node_update(float* __restrict__ h, const float* __restrict__ accum,
                 const float* __restrict__ denom, const float* __restrict__ skipb,
                 const float* __restrict__ g, const float* __restrict__ bta, int N) {
    int n = blockIdx.x;
    if (n >= N) return;
    int c = threadIdx.x;          // 0..127
    int lane = c & 31, wv = c >> 5;

    float val = accum[(size_t)n * 128 + c] /
                (denom[(size_t)n * HEADS + (c >> 5)] + SM_EPS) +
                skipb[(size_t)n * 128 + c];

    __shared__ float red[4];
    float s = val;
#pragma unroll
    for (int o = 16; o > 0; o >>= 1) s += __shfl_xor(s, o, 32);
    if (lane == 0) red[wv] = s;
    __syncthreads();
    float mu = (red[0] + red[1] + red[2] + red[3]) * (1.0f / 128.0f);
    __syncthreads();

    float dv = val - mu;
    float s2 = dv * dv;
#pragma unroll
    for (int o = 16; o > 0; o >>= 1) s2 += __shfl_xor(s2, o, 32);
    if (lane == 0) red[wv] = s2;
    __syncthreads();
    float var = (red[0] + red[1] + red[2] + red[3]) * (1.0f / 128.0f);

    float y = dv * rsqrtf(var + LN_EPS) * g[c] + bta[c];
    y = fmaxf(y, 0.0f);
    h[(size_t)n * 128 + c] += y;
}

// ---------------------------------------------------------------------------
// host orchestration
// ---------------------------------------------------------------------------
extern "C" void kernel_launch(void* const* d_in, const int* in_sizes, int n_in,
                              void* d_out, int out_size, void* d_ws, size_t ws_size,
                              hipStream_t stream) {
    (void)in_sizes; (void)n_in; (void)out_size; (void)ws_size;

    // -- inputs (setup_inputs dict order; params flattened depth-first) --
    const float*     xp      = (const float*)d_in[0];
    const float*     xs      = (const float*)d_in[1];
    const float*     xt      = (const float*)d_in[2];
    const long long* ps_ei   = (const long long*)d_in[3];
    const long long* st_ei   = (const long long*)d_in[4];
    const float*     ps_attr = (const float*)d_in[5];

    const float* pw[3] = {(const float*)d_in[6], (const float*)d_in[8],  (const float*)d_in[10]};
    const float* pb[3] = {(const float*)d_in[7], (const float*)d_in[9],  (const float*)d_in[11]};
    const float* ow    = (const float*)d_in[12];
    const float* ob    = (const float*)d_in[13];

    const float *qw[3], *qbs[3], *kw[3], *kbs[3], *vw[3], *vbs[3];
    const float *ew[3], *ebs[3], *sw[3], *sbs[3], *lg[3], *lb[3];
    for (int i = 0; i < 3; ++i) {
        int b = 14 + i * 12;
        qw[i]  = (const float*)d_in[b + 0];  qbs[i] = (const float*)d_in[b + 1];
        kw[i]  = (const float*)d_in[b + 2];  kbs[i] = (const float*)d_in[b + 3];
        vw[i]  = (const float*)d_in[b + 4];  vbs[i] = (const float*)d_in[b + 5];
        ew[i]  = (const float*)d_in[b + 6];  ebs[i] = (const float*)d_in[b + 7];
        sw[i]  = (const float*)d_in[b + 8];  sbs[i] = (const float*)d_in[b + 9];
        lg[i]  = (const float*)d_in[b + 10]; lb[i]  = (const float*)d_in[b + 11];
    }

    // -- workspace carve --
    char* ws = (char*)d_ws;
    size_t off = 0;
    auto carve = [&](size_t bytes) -> void* {
        void* p = ws + off;
        off += (bytes + 255) & ~(size_t)255;
        return p;
    };
    // packed bf16 weight arena: 3x(256*128) + 128*128 + 12x(128*128) ushorts
    const size_t OFF_PW = 0, OFF_SW = 32768, OFF_TW = 65536, OFF_OW = 98304, OFF_L0 = 114688;
    unsigned short* Wbf = (unsigned short*)carve((size_t)311296 * 2);
    float* hbuf  = (float*)carve((size_t)NTOT * 128 * 4);
    float* qb    = (float*)carve((size_t)NTOT * 128 * 4);
    float* kb    = (float*)carve((size_t)NTOT * 128 * 4);
    float* vb    = (float*)carve((size_t)NTOT * 128 * 4);
    float* skb   = (float*)carve((size_t)NTOT * 128 * 4);
    float* accum = (float*)carve((size_t)NTOT * 128 * 4);
    float* alphab= (float*)carve((size_t)ETOT * HEADS * 4);
    unsigned* amax = (unsigned*)carve((size_t)NTOT * HEADS * 4);
    float* denom = (float*)carve((size_t)NTOT * HEADS * 4);

    auto pack = [&](const float* src, size_t dstOff, int K) {
        int n = K * 128;
        pack_weight_bf16<<<dim3((n + 255) / 256), dim3(256), 0, stream>>>(src, Wbf + dstOff, K);
    };
    auto gemm1 = [&](const float* A, size_t wOff, const float* bias, float* C, int M, int K) {
        GemmArgs ga = {};
        ga.A = A; ga.M = M; ga.K = K; ga.nmat = 1;
        ga.W[0] = Wbf + wOff; ga.bias[0] = bias; ga.C[0] = C;
        gemm_multi<<<dim3((M + MTILE - 1) / MTILE), dim3(256), 0, stream>>>(ga);
    };

    // -- weights -> packed bf16 --
    pack(pw[0], OFF_PW, 256); pack(pw[1], OFF_SW, 256); pack(pw[2], OFF_TW, 256);
    pack(ow, OFF_OW, 128);
    for (int i = 0; i < 3; ++i) {
        size_t lo = OFF_L0 + (size_t)i * 65536;
        pack(qw[i], lo,          128);
        pack(kw[i], lo + 16384,  128);
        pack(vw[i], lo + 32768,  128);
        pack(sw[i], lo + 49152,  128);
    }

    // -- input projections into h --
    gemm1(xp, OFF_PW, pb[0], hbuf,                             NP_, D_IN);
    gemm1(xs, OFF_SW, pb[1], hbuf + (size_t)NP_ * 128,         NS_, D_IN);
    gemm1(xt, OFF_TW, pb[2], hbuf + (size_t)(NP_ + NS_) * 128, NT_, D_IN);

    // -- 3 TransformerConv layers --
    const int edgeGrid = (ETOT + 7) / 8;
    const int initGrid = (NTOT * 128 + 255) / 256;
    for (int L = 0; L < 3; ++L) {
        size_t lo = OFF_L0 + (size_t)L * 65536;

        // fused q/k/v/skip: stage h tile once, run 4 weight matrices against it
        GemmArgs ga = {};
        ga.A = hbuf; ga.M = NTOT; ga.K = D_H; ga.nmat = 4;
        ga.W[0] = Wbf + lo;         ga.bias[0] = qbs[L]; ga.C[0] = qb;
        ga.W[1] = Wbf + lo + 16384; ga.bias[1] = kbs[L]; ga.C[1] = kb;
        ga.W[2] = Wbf + lo + 32768; ga.bias[2] = vbs[L]; ga.C[2] = vb;
        ga.W[3] = Wbf + lo + 49152; ga.bias[3] = sbs[L]; ga.C[3] = skb;
        gemm_multi<<<dim3((NTOT + MTILE - 1) / MTILE), dim3(256), 0, stream>>>(ga);

        init_layer<<<dim3(initGrid), dim3(256), 0, stream>>>(
            accum, denom, amax, NTOT * 128, NTOT * HEADS);

        edge_pass1<<<dim3(edgeGrid), dim3(256), 0, stream>>>(
            qb, kb, ps_ei, st_ei, ps_attr, ew[L], ebs[L], alphab, amax);

        edge_pass2<<<dim3(edgeGrid), dim3(256), 0, stream>>>(
            vb, ps_ei, st_ei, ps_attr, ew[L], ebs[L], alphab, amax, denom, accum);

        node_update<<<dim3(NTOT), dim3(128), 0, stream>>>(
            hbuf, accum, denom, skb, lg[L], lb[L], NTOT);
    }

    // -- output projection --
    gemm1(hbuf, OFF_OW, ob, (float*)d_out, NTOT, D_H);
}
